// GraphSAGE_11879879540745
// MI455X (gfx1250) — matrix operand order repaired and checked by reading, not compiled, not run on hardware
//
#include <hip/hip_runtime.h>

// ---------------------------------------------------------------------------
// GraphSAGE forward for MI455X (gfx1250, wave32, WMMA).
// Activations: bf16.  Edge aggregation + WMMA accumulation: fp32.
// ---------------------------------------------------------------------------

typedef __attribute__((ext_vector_type(8)))  __bf16 v8bf;
typedef __attribute__((ext_vector_type(16))) __bf16 v16bf;
typedef __attribute__((ext_vector_type(8)))  float  v8f;

#define NODES  65536
#define EDGES  524288
#define HDIM   256
#define IN_DIM 128
#define BROWS  2048      // B * H = 8 * 256
#define MDIM   8192      // MAXN

// ---------------------------------------------------------------------------
// Weight prep: WcatT[n][k] = (k<din ? Wl[k][n] : Wr[k-din][n]) as bf16
// ---------------------------------------------------------------------------
__global__ void prep_wcat(const float* __restrict__ Wl, const float* __restrict__ Wr,
                          __bf16* __restrict__ dst, int din, int dout) {
  int i = blockIdx.x * 256 + threadIdx.x;
  int K2 = 2 * din;
  if (i >= dout * K2) return;
  int n = i / K2, k = i - n * K2;
  float v = (k < din) ? Wl[(size_t)k * dout + n] : Wr[(size_t)(k - din) * dout + n];
  dst[(size_t)n * K2 + k] = (__bf16)v;
}

// WT[n][k] = W[k][n] as bf16
__global__ void prep_wt(const float* __restrict__ W, __bf16* __restrict__ dst,
                        int K, int Nn) {
  int i = blockIdx.x * 256 + threadIdx.x;
  if (i >= K * Nn) return;
  int n = i / K, k = i - n * K;
  dst[(size_t)n * K + k] = (__bf16)W[(size_t)k * Nn + n];
}

// h0 = concat(x, g0, g1, g2) -> bf16 [NODES, 128]
__global__ void concat_kernel(const float* __restrict__ x, const float* __restrict__ g0,
                              const float* __restrict__ g1, const float* __restrict__ g2,
                              __bf16* __restrict__ out) {
  int i = blockIdx.x * 256 + threadIdx.x;
  if (i >= NODES * IN_DIM) return;
  int r = i >> 7, c = i & 127;
  float v;
  if      (c < 32)  v = x [(size_t)r * 32 + c];
  else if (c < 64)  v = g0[(size_t)r * 32 + (c - 32)];
  else if (c < 96)  v = g1[(size_t)r * 32 + (c - 64)];
  else              v = g2[(size_t)r * 32 + (c - 96)];
  out[i] = (__bf16)v;
}

__global__ void zero_f32(float* __restrict__ p, int n4) {
  int i = blockIdx.x * 256 + threadIdx.x;
  if (i < n4) ((float4*)p)[i] = make_float4(0.f, 0.f, 0.f, 0.f);
}

__global__ void deg_kernel(const int* __restrict__ edst, float* __restrict__ cnt) {
  int e = blockIdx.x * 256 + threadIdx.x;
  if (e < EDGES) atomicAdd(cnt + edst[e], 1.0f);
}

__global__ void rcnt_kernel(float* __restrict__ cnt) {
  int i = blockIdx.x * 256 + threadIdx.x;
  if (i < NODES) cnt[i] = 1.0f / fmaxf(cnt[i], 1.0f);
}

// agg[dst] += h[src] * rcnt[dst]  (mean folded into the add).
// Thread layout: lane = feature-chunk (8 bf16), wave spans one full row.
__global__ void __launch_bounds__(256) scatter_mean(
    const __bf16* __restrict__ h, const int* __restrict__ esrc,
    const int* __restrict__ edst, const float* __restrict__ rcnt,
    float* __restrict__ agg, int chunksLog2, int din) {
  int t = blockIdx.x * 256 + threadIdx.x;
  int c = t & ((1 << chunksLog2) - 1);
  int e = t >> chunksLog2;
  if (e >= EDGES) return;
  int s = esrc[e], d = edst[e];
  float r = rcnt[d];
  v8bf hv = *(const v8bf*)(h + (size_t)s * din + c * 8);
  float* ap = agg + (size_t)d * din + c * 8;
#pragma unroll
  for (int i = 0; i < 8; i++) atomicAdd(ap + i, (float)hv[i] * r);
}

// ---------------------------------------------------------------------------
// Fragment loaders.
// A-fragment lane layout (matches B): lane<16: M=lane, K=k0..k0+15 contiguous;
//   lane>=16: M=lane-16, K=k0+16..k0+31.
// B-fragment lane layout (same k order as A): lane<16: N=lane, K=k0..k0+15
//   contiguous; lane>=16: N=lane-16, K=k0+16..k0+31.
// ---------------------------------------------------------------------------
__device__ __forceinline__ v16bf ldfrag(const __bf16* p) {
  v8bf lo = *(const v8bf*)(p);
  v8bf hi = *(const v8bf*)(p + 8);
  return __builtin_shufflevector(lo, hi, 0,1,2,3,4,5,6,7,8,9,10,11,12,13,14,15);
}

__device__ __forceinline__ v16bf cvtfrag(const float* ap) {
  float4 f0 = *(const float4*)(ap);
  float4 f1 = *(const float4*)(ap + 4);
  float4 f2 = *(const float4*)(ap + 8);
  float4 f3 = *(const float4*)(ap + 12);
  v16bf a;
  a[0]  = (__bf16)f0.x; a[1]  = (__bf16)f0.y; a[2]  = (__bf16)f0.z; a[3]  = (__bf16)f0.w;
  a[4]  = (__bf16)f1.x; a[5]  = (__bf16)f1.y; a[6]  = (__bf16)f1.z; a[7]  = (__bf16)f1.w;
  a[8]  = (__bf16)f2.x; a[9]  = (__bf16)f2.y; a[10] = (__bf16)f2.z; a[11] = (__bf16)f2.w;
  a[12] = (__bf16)f3.x; a[13] = (__bf16)f3.y; a[14] = (__bf16)f3.z; a[15] = (__bf16)f3.w;
  return a;
}

#define WMMA_BF16(A, B, C) \
  __builtin_amdgcn_wmma_f32_16x16x32_bf16(false, (A), false, (B), (short)0, (C), false, false)

// ---------------------------------------------------------------------------
// WMMA GEMM:  C = relu( [bf16(aggF32) | Abf] @ WT^T + bias ),  C in bf16.
//   A is [M, K2]: cols [0,din) from fp32 agg (converted in-register),
//   cols [din,K2) from bf16 Abf (width K2-din).  WT is [Nn][K2] bf16 row-major
//   (pre-transposed), bias [Nn] fp32.
// Block = 256 threads = 8 waves; block tile 256(M) x 64(N); wave tile 32x64:
//   2 A-frags reuse 4 B-frags -> 8 WMMAs per 6 fragment loads.
// The bf16 K-phase is software-pipelined (next fragments preloaded in regs)
// so v_wmma issues overlap global-load latency instead of s_wait'ing on it.
// Requires: M%256==0, Nn%64==0, din%32==0, (K2-din)>=64 and %32==0.
// ---------------------------------------------------------------------------
__global__ void __launch_bounds__(256) wmma_gemm_bf16(
    const float* __restrict__ aggF32, const __bf16* __restrict__ Abf,
    const __bf16* __restrict__ WT, const float* __restrict__ bias,
    __bf16* __restrict__ Cout, int din, int K2, int Nn) {
  const int wave  = threadIdx.x >> 5;
  const int lane  = threadIdx.x & 31;
  const int lrow  = lane & 15;
  const int hi16  = lane >> 4;                 // 0 or 1
  const int mBase = blockIdx.x * 256 + wave * 32;
  const int nBase = blockIdx.y * 64;
  const int m0    = mBase + lrow;
  const int dinB  = K2 - din;                  // width of bf16 part of A
  const int kA    = hi16 * 16;                 // A contiguous half
  const int kB    = hi16 * 16;                 // B contiguous half

  v8f c0 = {}, c1 = {}, c2 = {}, c3 = {}, c4 = {}, c5 = {}, c6 = {}, c7 = {};

  const __bf16* bRow = WT + (size_t)(nBase + lrow) * K2 + kB;
  const size_t  bS   = (size_t)16 * K2;        // 16 columns of WT

  // ---- phase 1: fp32 agg columns [0, din)  (short: <= 8 iterations) -------
  for (int k0 = 0; k0 < din; k0 += 32) {
    v16bf a0 = cvtfrag(aggF32 + (size_t)m0 * din + k0 + kA);
    v16bf a1 = cvtfrag(aggF32 + (size_t)(m0 + 16) * din + k0 + kA);
    v16bf b0 = *(const v16bf*)(bRow + k0);
    v16bf b1 = *(const v16bf*)(bRow + k0 + bS);
    v16bf b2 = *(const v16bf*)(bRow + k0 + 2 * bS);
    v16bf b3 = *(const v16bf*)(bRow + k0 + 3 * bS);
    c0 = WMMA_BF16(a0, b0, c0); c1 = WMMA_BF16(a0, b1, c1);
    c2 = WMMA_BF16(a0, b2, c2); c3 = WMMA_BF16(a0, b3, c3);
    c4 = WMMA_BF16(a1, b0, c4); c5 = WMMA_BF16(a1, b1, c5);
    c6 = WMMA_BF16(a1, b2, c6); c7 = WMMA_BF16(a1, b3, c7);
  }

  // ---- phase 2: bf16 self columns [din, K2), software-pipelined -----------
  {
    const __bf16* a0p = Abf + (size_t)m0 * dinB + kA;
    const __bf16* a1p = Abf + (size_t)(m0 + 16) * dinB + kA;
    const __bf16* bp  = bRow + din;
    v16bf a0 = ldfrag(a0p);
    v16bf a1 = ldfrag(a1p);
    v16bf b0 = *(const v16bf*)(bp);
    v16bf b1 = *(const v16bf*)(bp + bS);
    v16bf b2 = *(const v16bf*)(bp + 2 * bS);
    v16bf b3 = *(const v16bf*)(bp + 3 * bS);
    int kk = 0;
    for (; kk < dinB - 32; kk += 32) {
      // issue next-step loads before the WMMAs so they overlap matrix ops
      v16bf na0 = ldfrag(a0p + kk + 32);
      v16bf na1 = ldfrag(a1p + kk + 32);
      v16bf nb0 = *(const v16bf*)(bp + kk + 32);
      v16bf nb1 = *(const v16bf*)(bp + kk + 32 + bS);
      v16bf nb2 = *(const v16bf*)(bp + kk + 32 + 2 * bS);
      v16bf nb3 = *(const v16bf*)(bp + kk + 32 + 3 * bS);
      __builtin_prefetch((const void*)(a0p + kk + 64), 0, 1);  // global_prefetch_b8
      c0 = WMMA_BF16(a0, b0, c0); c1 = WMMA_BF16(a0, b1, c1);
      c2 = WMMA_BF16(a0, b2, c2); c3 = WMMA_BF16(a0, b3, c3);
      c4 = WMMA_BF16(a1, b0, c4); c5 = WMMA_BF16(a1, b1, c5);
      c6 = WMMA_BF16(a1, b2, c6); c7 = WMMA_BF16(a1, b3, c7);
      a0 = na0; a1 = na1; b0 = nb0; b1 = nb1; b2 = nb2; b3 = nb3;
    }
    // peeled last step
    c0 = WMMA_BF16(a0, b0, c0); c1 = WMMA_BF16(a0, b1, c1);
    c2 = WMMA_BF16(a0, b2, c2); c3 = WMMA_BF16(a0, b3, c3);
    c4 = WMMA_BF16(a1, b0, c4); c5 = WMMA_BF16(a1, b1, c5);
    c6 = WMMA_BF16(a1, b2, c6); c7 = WMMA_BF16(a1, b3, c7);
  }

  // ---- Epilogue: C/D layout (ISA 7.12.2): VGPR v, lane<16 -> M=v, lane>=16 -> M=v+8
  const int col = nBase + lrow;
  float bb0 = bias[col], bb1 = bias[col + 16], bb2 = bias[col + 32], bb3 = bias[col + 48];
#pragma unroll
  for (int v = 0; v < 8; v++) {
    int row = mBase + v + hi16 * 8;
    size_t base = (size_t)row * Nn;
    Cout[base + col     ] = (__bf16)fmaxf(c0[v] + bb0, 0.f);
    Cout[base + col + 16] = (__bf16)fmaxf(c1[v] + bb1, 0.f);
    Cout[base + col + 32] = (__bf16)fmaxf(c2[v] + bb2, 0.f);
    Cout[base + col + 48] = (__bf16)fmaxf(c3[v] + bb3, 0.f);
    size_t base2 = base + (size_t)16 * Nn;
    Cout[base2 + col     ] = (__bf16)fmaxf(c4[v] + bb0, 0.f);
    Cout[base2 + col + 16] = (__bf16)fmaxf(c5[v] + bb1, 0.f);
    Cout[base2 + col + 32] = (__bf16)fmaxf(c6[v] + bb2, 0.f);
    Cout[base2 + col + 48] = (__bf16)fmaxf(c7[v] + bb3, 0.f);
  }
}

// ---------------------------------------------------------------------------
// Tail: mlp1 @W2 reduction, batchnorm (B=8), mlp2 — tiny, fp32 VALU.
// ---------------------------------------------------------------------------
__global__ void mlp1_out(const __bf16* __restrict__ h5, const float* __restrict__ W2,
                         const float* __restrict__ b2, float* __restrict__ h6) {
  int r = blockIdx.x * 256 + threadIdx.x;
  if (r >= BROWS) return;
  float acc = b2[0];
  const __bf16* hp = h5 + (size_t)r * HDIM;
  for (int k = 0; k < HDIM; k++) acc += (float)hp[k] * W2[k];
  h6[r] = acc;
}

__global__ void __launch_bounds__(256) bn_kernel(const float* __restrict__ h6,
                                                 const float* __restrict__ gamma,
                                                 const float* __restrict__ beta,
                                                 float* __restrict__ h7) {
  int t = threadIdx.x;
  float vals[8], mean = 0.f;
#pragma unroll
  for (int b = 0; b < 8; b++) { vals[b] = h6[b * HDIM + t]; mean += vals[b]; }
  mean *= 0.125f;
  float var = 0.f;
#pragma unroll
  for (int b = 0; b < 8; b++) { float d = vals[b] - mean; var += d * d; }
  var *= 0.125f;                               // population var (jnp.var ddof=0)
  float sc = gamma[t] * rsqrtf(var + 1e-5f);
  float sh = beta[t];
#pragma unroll
  for (int b = 0; b < 8; b++) h7[b * HDIM + t] = fmaxf((vals[b] - mean) * sc + sh, 0.f);
}

__global__ void mlp2a(const float* __restrict__ h7, const float* __restrict__ W1,
                      const float* __restrict__ b1, float* __restrict__ h8) {
  int t = blockIdx.x * 256 + threadIdx.x;     // 0..2047
  int row = t >> 8, col = t & 255;
  float acc = b1[col];
  for (int k = 0; k < HDIM; k++) acc += h7[row * HDIM + k] * W1[(size_t)k * HDIM + col];
  h8[row * HDIM + col] = fmaxf(acc, 0.f);
}

__global__ void mlp2b(const float* __restrict__ h8, const float* __restrict__ W2,
                      const float* __restrict__ b2, float* __restrict__ out) {
  int t = threadIdx.x;
  if (t >= 64) return;
  int row = t >> 3, o = t & 7;
  float acc = b2[o];
  for (int c = 0; c < HDIM; c++) acc += h8[row * HDIM + c] * W2[c * 8 + o];
  out[row * 8 + o] = acc;
}

// ---------------------------------------------------------------------------
extern "C" void kernel_launch(void* const* d_in, const int* in_sizes, int n_in,
                              void* d_out, int out_size, void* d_ws, size_t ws_size,
                              hipStream_t stream) {
  (void)in_sizes; (void)n_in; (void)out_size; (void)ws_size;
  const float* x    = (const float*)d_in[0];
  const float* g0   = (const float*)d_in[1];
  const float* g1   = (const float*)d_in[2];
  const float* g2   = (const float*)d_in[3];
  const int*   esrc = (const int*)  d_in[4];
  const int*   edst = (const int*)  d_in[5];
  const float* Wl[4]  = {(const float*)d_in[6],  (const float*)d_in[9],
                         (const float*)d_in[12], (const float*)d_in[15]};
  const float* Wr[4]  = {(const float*)d_in[7],  (const float*)d_in[10],
                         (const float*)d_in[13], (const float*)d_in[16]};
  const float* bb[4]  = {(const float*)d_in[8],  (const float*)d_in[11],
                         (const float*)d_in[14], (const float*)d_in[17]};
  const float* m1W1 = (const float*)d_in[18];
  const float* m1b1 = (const float*)d_in[19];
  const float* m1W2 = (const float*)d_in[20];
  const float* m1b2 = (const float*)d_in[21];
  const float* gam  = (const float*)d_in[22];
  const float* bet  = (const float*)d_in[23];
  const float* m2W1 = (const float*)d_in[24];
  const float* m2b1 = (const float*)d_in[25];
  const float* m2W2 = (const float*)d_in[26];
  const float* m2b2 = (const float*)d_in[27];

  char* ws = (char*)d_ws;
  size_t off = 0;
  auto alloc = [&](size_t bytes) { size_t o = off; off += (bytes + 255) & ~(size_t)255; return o; };
  __bf16* bufA = (__bf16*)(ws + alloc((size_t)NODES * HDIM * 2));
  __bf16* bufB = (__bf16*)(ws + alloc((size_t)NODES * HDIM * 2));
  float*  agg  = (float*) (ws + alloc((size_t)NODES * HDIM * 4));
  float*  rcnt = (float*) (ws + alloc((size_t)NODES * 4));
  __bf16* WT[4];
  WT[0] = (__bf16*)(ws + alloc((size_t)HDIM * 256 * 2));
  for (int l = 1; l < 4; l++) WT[l] = (__bf16*)(ws + alloc((size_t)HDIM * 512 * 2));
  __bf16* W1T = (__bf16*)(ws + alloc((size_t)HDIM * MDIM * 2));
  float*  h6  = (float*) (ws + alloc((size_t)BROWS * 4));
  float*  h7  = (float*) (ws + alloc((size_t)8 * HDIM * 4));
  float*  h8  = (float*) (ws + alloc((size_t)8 * HDIM * 4));

  // --- weight prep (bf16 transposed) ---
  prep_wcat<<<(HDIM * 256 + 255) / 256, 256, 0, stream>>>(Wl[0], Wr[0], WT[0], 128, HDIM);
  for (int l = 1; l < 4; l++)
    prep_wcat<<<(HDIM * 512 + 255) / 256, 256, 0, stream>>>(Wl[l], Wr[l], WT[l], 256, HDIM);
  prep_wt<<<(MDIM * HDIM + 255) / 256, 256, 0, stream>>>(m1W1, W1T, MDIM, HDIM);

  // --- input concat + degree ---
  concat_kernel<<<(NODES * IN_DIM + 255) / 256, 256, 0, stream>>>(x, g0, g1, g2, bufA);
  zero_f32<<<(NODES / 4 + 255) / 256, 256, 0, stream>>>(rcnt, NODES / 4);
  deg_kernel<<<EDGES / 256, 256, 0, stream>>>(edst, rcnt);
  rcnt_kernel<<<NODES / 256, 256, 0, stream>>>(rcnt);

  // --- 4 SAGEConv layers ---
  __bf16* cur = bufA;
  __bf16* nxt = bufB;
  int din = IN_DIM;
  for (int l = 0; l < 4; l++) {
    int n4 = NODES * din / 4;
    zero_f32<<<(n4 + 255) / 256, 256, 0, stream>>>(agg, n4);
    int clog = (din == 128) ? 4 : 5;                    // din/8 chunks per edge
    long tot = (long)EDGES << clog;
    scatter_mean<<<(int)(tot / 256), 256, 0, stream>>>(cur, esrc, edst, rcnt, agg, clog, din);
    dim3 grid(NODES / 256, HDIM / 64);
    wmma_gemm_bf16<<<grid, 256, 0, stream>>>(agg, cur, WT[l], bb[l], nxt, din, 2 * din, HDIM);
    __bf16* t2 = cur; cur = nxt; nxt = t2;
    din = HDIM;
  }

  // --- mlp1: [2048, 8192] @ [8192, 256] (+bias, relu); A = cur reinterpreted ---
  {
    dim3 grid(BROWS / 256, HDIM / 64);
    wmma_gemm_bf16<<<grid, 256, 0, stream>>>(nullptr, cur, W1T, m1b1, nxt, 0, MDIM, HDIM);
  }
  mlp1_out<<<BROWS / 256, 256, 0, stream>>>(nxt, m1W2, m1b2, h6);
  bn_kernel<<<1, 256, 0, stream>>>(h6, gam, bet, h7);
  mlp2a<<<8, 256, 0, stream>>>(h7, m2W1, m2b1, h8);
  mlp2b<<<1, 64, 0, stream>>>(h8, m2W2, m2b2, (float*)d_out);
}
